// MultiHead_14998025798074
// MI455X (gfx1250) — matrix operand order repaired and checked by reading, not compiled, not run on hardware
//
#include <hip/hip_runtime.h>
#include <hip/hip_bf16.h>

// ---------------------------------------------------------------------------
// MultiHead attention for MI455X (gfx1250), wave32 + v_wmma_f32_16x16x32_f16.
// B=2, S=2048, D=512, H=8, dh=64.  Flash-attention with causal masking.
// Weights are converted fp32->f16 once; activations convert in-register.
// ---------------------------------------------------------------------------

typedef __attribute__((ext_vector_type(16))) _Float16 v16h;
typedef __attribute__((ext_vector_type(8)))  float    v8f;

#define SQ     2048
#define DK     512
#define NH     8
#define DH     64
#define NB     2
#define MROWS  (NB * SQ)          // 4096
#define WELT   (DK * DK)          // 262144 elements per weight matrix
#define INFC   10000000000.0f
#define RSCALE 0.04419417382415922f   // 1/sqrt(512)

union Frag16 { v16h v; _Float16 e[16]; uint4 u[2]; };

// Load a 16x32 f16 A-fragment or 32x16 f16 B-fragment given the two 16-byte
// contiguous runs this lane owns (per ISA 7.12.2 layouts).
__device__ __forceinline__ v16h load_frag_f16(const _Float16* p0, const _Float16* p1) {
  Frag16 f;
  f.u[0] = *(const uint4*)p0;
  f.u[1] = *(const uint4*)p1;
  return f.v;
}

// Build an f16 fragment from two contiguous 8-float runs (fp32 source),
// converting in-register (v_cvt_pk_f16_f32).
__device__ __forceinline__ v16h cvt_frag_f32(const float* p0, const float* p1) {
  Frag16 f;
  float4 a = *(const float4*)(p0);
  float4 b = *(const float4*)(p0 + 4);
  float4 c = *(const float4*)(p1);
  float4 d = *(const float4*)(p1 + 4);
  f.e[0]  = (_Float16)a.x; f.e[1]  = (_Float16)a.y;
  f.e[2]  = (_Float16)a.z; f.e[3]  = (_Float16)a.w;
  f.e[4]  = (_Float16)b.x; f.e[5]  = (_Float16)b.y;
  f.e[6]  = (_Float16)b.z; f.e[7]  = (_Float16)b.w;
  f.e[8]  = (_Float16)c.x; f.e[9]  = (_Float16)c.y;
  f.e[10] = (_Float16)c.z; f.e[11] = (_Float16)c.w;
  f.e[12] = (_Float16)d.x; f.e[13] = (_Float16)d.y;
  f.e[14] = (_Float16)d.z; f.e[15] = (_Float16)d.w;
  return f.v;
}

// ---------------------------------------------------------------------------
// One-pass fp32 -> f16 weight conversion (row-major preserved).
// 256 blocks x 256 threads, 4 elements per thread.
// ---------------------------------------------------------------------------
__global__ __launch_bounds__(256) void cvt_w_kernel(
    const float* __restrict__ w, _Float16* __restrict__ w16) {
  int i = (blockIdx.x * 256 + threadIdx.x) * 4;
  float4 f = *(const float4*)(w + i);
  union { _Float16 h[4]; uint2 u; } o;
  o.h[0] = (_Float16)f.x; o.h[1] = (_Float16)f.y;
  o.h[2] = (_Float16)f.z; o.h[3] = (_Float16)f.w;
  *(uint2*)(w16 + i) = o.u;
}

// ---------------------------------------------------------------------------
// Projection: out = x @ W^T (nn.Linear), x [4096,512] fp32, W16 [512,512] f16.
// Each wave computes a 16x64 output tile (A reused across 4 WMMAs).
// layout 0: out[b][h][s][d]  (Q, K)     layout 1: out[b][h][d][s] (V^T)
// ---------------------------------------------------------------------------
__global__ __launch_bounds__(256) void proj_kernel(
    const float* __restrict__ x, const _Float16* __restrict__ w16,
    _Float16* __restrict__ out, int layout) {
  const int wid   = (blockIdx.x * blockDim.x + threadIdx.x) >> 5;  // 0..2047
  const int lane  = threadIdx.x & 31;
  const int nl    = lane & 15;
  const int g     = lane >> 4;
  const int mbase = (wid >> 3) * 16;   // 256 M-tiles
  const int nbase = (wid & 7) * 64;    // 8 N-groups of 64

  v8f acc[4] = {};
  const float* xrow = x + (mbase + nl) * DK;   // A row = lane&15
  for (int k0 = 0; k0 < DK; k0 += 32) {
    const float* pa = xrow + k0 + g * 8;       // K runs [g*8, +8) and +16
    v16h afrag = cvt_frag_f32(pa, pa + 16);
#pragma unroll
    for (int j = 0; j < 4; ++j) {
      const _Float16* pb = w16 + (nbase + j * 16 + nl) * DK + k0 + g * 8;
      v16h bfrag = load_frag_f16(pb, pb + 16);  // K runs [g*8, +8) and +16 for B
      acc[j] = __builtin_amdgcn_wmma_f32_16x16x32_f16(
          false, afrag, false, bfrag, (short)0, acc[j], false, false);
    }
  }
#pragma unroll
  for (int j = 0; j < 4; ++j) {
#pragma unroll
    for (int r = 0; r < 8; ++r) {
      int row = mbase + g * 8 + r;             // D-frag: m = g*8+r, n = lane&15
      int col = nbase + j * 16 + nl;
      int b = row >> 11, s = row & (SQ - 1);
      int h = col >> 6,  d = col & (DH - 1);
      int idx = (layout == 0)
          ? (((b * NH + h) * SQ + s) * DH + d)
          : (((b * NH + h) * DH + d) * SQ + s);
      out[idx] = (_Float16)acc[j][r];
    }
  }
}

// ---------------------------------------------------------------------------
// Flash attention: one wave per (b, head, 16-query tile). Streams 32-key tiles
// with causal masking, online softmax, P staged through LDS for the PV GEMM.
// ---------------------------------------------------------------------------
__global__ __launch_bounds__(256) void attn_kernel(
    const _Float16* __restrict__ Qb,   // [B,H,S,64]
    const _Float16* __restrict__ Kb,   // [B,H,S,64]
    const _Float16* __restrict__ Vt,   // [B,H,64,S]
    _Float16* __restrict__ attn) {     // [B,S,512]
  __shared__ __align__(16) _Float16 pbuf[8][16 * 32];   // 1KB per wave
  const int wib   = threadIdx.x >> 5;
  const int lane  = threadIdx.x & 31;
  const int gwid  = blockIdx.x * 8 + wib;               // 0..2047
  const int qtile = gwid & 127;
  const int head  = (gwid >> 7) & (NH - 1);
  const int b     = gwid >> 10;
  const int qbase = qtile * 16;
  const int nl    = lane & 15;
  const int g     = lane >> 4;

  const _Float16* Qh = Qb + ((b * NH + head) * SQ) * DH;
  const _Float16* Kh = Kb + ((b * NH + head) * SQ) * DH;
  const _Float16* Vh = Vt + ((b * NH + head) * DH) * SQ;

  // Preload the two Q A-fragments (d = 0..31, 32..63); reused all iterations.
  v16h qf[2];
  {
    const _Float16* qrow = Qh + (qbase + nl) * DH;
#pragma unroll
    for (int ks = 0; ks < 2; ++ks) {
      const _Float16* p = qrow + ks * 32 + g * 8;
      qf[ks] = load_frag_f16(p, p + 16);
    }
  }

  v8f acc[4] = {};
  float mrow[8], lrow[8];
#pragma unroll
  for (int r = 0; r < 8; ++r) { mrow[r] = -3.0e38f; lrow[r] = 0.0f; }

  const int kvend = ((qbase + 15) >> 5) << 5;   // last causal kv tile base
  for (int kvb = 0; kvb <= kvend; kvb += 32) {
    if (kvb < kvend) {   // prefetch next K tile rows (global_prefetch_b8)
      __builtin_prefetch(Kh + (kvb + 32 + nl) * DH, 0, 0);
      __builtin_prefetch(Kh + (kvb + 48 + nl) * DH, 0, 0);
    }
    // ---- S = Q K^T for two 16-key subtiles -------------------------------
    v8f sfr[2];
#pragma unroll
    for (int t = 0; t < 2; ++t) {
      v8f c = {};
#pragma unroll
      for (int ks = 0; ks < 2; ++ks) {
        const _Float16* p = Kh + (kvb + t * 16 + nl) * DH + ks * 32 + g * 8;
        v16h bf = load_frag_f16(p, p + 16);
        c = __builtin_amdgcn_wmma_f32_16x16x32_f16(
            false, qf[ks], false, bf, (short)0, c, false, false);
      }
      sfr[t] = c;
    }
    // ---- causal mask (subtract INF pre-scale, per reference) + softmax ---
    float p0[8], p1[8];
#pragma unroll
    for (int r = 0; r < 8; ++r) {
      int qg = qbase + g * 8 + r;
      float s0 = sfr[0][r] - ((kvb + nl)      > qg ? INFC : 0.0f);
      float s1 = sfr[1][r] - ((kvb + 16 + nl) > qg ? INFC : 0.0f);
      s0 *= RSCALE; s1 *= RSCALE;
      float mx = fmaxf(s0, s1);
      mx = fmaxf(mx, __shfl_xor(mx, 1, 32));
      mx = fmaxf(mx, __shfl_xor(mx, 2, 32));
      mx = fmaxf(mx, __shfl_xor(mx, 4, 32));
      mx = fmaxf(mx, __shfl_xor(mx, 8, 32));
      float mnew  = fmaxf(mrow[r], mx);
      float alpha = __expf(mrow[r] - mnew);
      float e0 = __expf(s0 - mnew), e1 = __expf(s1 - mnew);
      float rs = e0 + e1;
      rs += __shfl_xor(rs, 1, 32);
      rs += __shfl_xor(rs, 2, 32);
      rs += __shfl_xor(rs, 4, 32);
      rs += __shfl_xor(rs, 8, 32);
      lrow[r] = lrow[r] * alpha + rs;
      mrow[r] = mnew;
      p0[r] = e0; p1[r] = e1;
#pragma unroll
      for (int j = 0; j < 4; ++j) acc[j][r] *= alpha;
    }
    // ---- D-fragment -> A-fragment transpose of P through LDS -------------
    _Float16* pl = &pbuf[wib][0];
#pragma unroll
    for (int r = 0; r < 8; ++r) {
      int m = g * 8 + r;
      pl[m * 32 + nl]      = (_Float16)p0[r];
      pl[m * 32 + 16 + nl] = (_Float16)p1[r];
    }
    asm volatile("s_wait_dscnt 0" ::: "memory");  // wave-local LDS RAW fence
    Frag16 pf;
    pf.u[0] = *(const uint4*)(&pbuf[wib][nl * 32 + g * 8]);
    pf.u[1] = *(const uint4*)(&pbuf[wib][nl * 32 + g * 8 + 16]);
    // ---- acc += P @ V  (V stored transposed: contiguous kv per d-row) ----
#pragma unroll
    for (int j = 0; j < 4; ++j) {
      const _Float16* p = Vh + (j * 16 + nl) * SQ + kvb + g * 8;
      v16h vf = load_frag_f16(p, p + 16);
      acc[j] = __builtin_amdgcn_wmma_f32_16x16x32_f16(
          false, pf.v, false, vf, (short)0, acc[j], false, false);
    }
  }
  // ---- normalize and emit f16 attn [B,S,512] -----------------------------
#pragma unroll
  for (int j = 0; j < 4; ++j) {
#pragma unroll
    for (int r = 0; r < 8; ++r) {
      int row = qbase + g * 8 + r;
      float o = acc[j][r] / lrow[r];
      attn[(b * SQ + row) * DK + head * DH + j * 16 + nl] = (_Float16)o;
    }
  }
}

// ---------------------------------------------------------------------------
// Output projection: out = attn @ wo^T.  A and B both f16 (direct b128
// fragment loads), fp32 output to d_out.
// ---------------------------------------------------------------------------
__global__ __launch_bounds__(256) void oproj_kernel(
    const _Float16* __restrict__ attn, const _Float16* __restrict__ wo16,
    float* __restrict__ out) {
  const int wid   = (blockIdx.x * blockDim.x + threadIdx.x) >> 5;
  const int lane  = threadIdx.x & 31;
  const int nl    = lane & 15;
  const int g     = lane >> 4;
  const int mbase = (wid >> 3) * 16;
  const int nbase = (wid & 7) * 64;

  v8f acc[4] = {};
  const _Float16* arow = attn + (mbase + nl) * DK;
  for (int k0 = 0; k0 < DK; k0 += 32) {
    const _Float16* pa = arow + k0 + g * 8;
    v16h afrag = load_frag_f16(pa, pa + 16);
#pragma unroll
    for (int j = 0; j < 4; ++j) {
      const _Float16* pb = wo16 + (nbase + j * 16 + nl) * DK + k0 + g * 8;
      v16h bfrag = load_frag_f16(pb, pb + 16);
      acc[j] = __builtin_amdgcn_wmma_f32_16x16x32_f16(
          false, afrag, false, bfrag, (short)0, acc[j], false, false);
    }
  }
#pragma unroll
  for (int j = 0; j < 4; ++j) {
#pragma unroll
    for (int r = 0; r < 8; ++r) {
      int row = mbase + g * 8 + r;
      int col = nbase + j * 16 + nl;
      out[row * DK + col] = acc[j][r];
    }
  }
}

// ---------------------------------------------------------------------------
extern "C" void kernel_launch(void* const* d_in, const int* in_sizes, int n_in,
                              void* d_out, int out_size, void* d_ws, size_t ws_size,
                              hipStream_t stream) {
  const float* q  = (const float*)d_in[0];
  const float* k  = (const float*)d_in[1];
  const float* v  = (const float*)d_in[2];
  const float* wq = (const float*)d_in[3];
  const float* wk = (const float*)d_in[4];
  const float* wv = (const float*)d_in[5];
  const float* wo = (const float*)d_in[6];
  float* out = (float*)d_out;

  const size_t ELT = (size_t)MROWS * DK;     // 2,097,152 f16 per buffer
  _Float16* Qb   = (_Float16*)d_ws;
  _Float16* Kb   = Qb + ELT;
  _Float16* Vt   = Kb + ELT;
  _Float16* At   = Vt + ELT;
  _Float16* Wq16 = At + ELT;                 // 4 x 512KB f16 weight copies
  _Float16* Wk16 = Wq16 + WELT;
  _Float16* Wv16 = Wk16 + WELT;
  _Float16* Wo16 = Wv16 + WELT;              // total 18 MB of workspace

  dim3 blk(256), grd(256);                   // 2048 wave32s per GEMM stage
  cvt_w_kernel<<<grd, blk, 0, stream>>>(wq, Wq16);
  cvt_w_kernel<<<grd, blk, 0, stream>>>(wk, Wk16);
  cvt_w_kernel<<<grd, blk, 0, stream>>>(wv, Wv16);
  cvt_w_kernel<<<grd, blk, 0, stream>>>(wo, Wo16);
  proj_kernel<<<grd, blk, 0, stream>>>(q, Wq16, Qb, 0);
  proj_kernel<<<grd, blk, 0, stream>>>(k, Wk16, Kb, 0);
  proj_kernel<<<grd, blk, 0, stream>>>(v, Wv16, Vt, 1); // V stored transposed
  attn_kernel<<<grd, blk, 0, stream>>>(Qb, Kb, Vt, At);
  oproj_kernel<<<grd, blk, 0, stream>>>(At, Wo16, out);
}
